// Attention_42013370090020
// MI455X (gfx1250) — matrix operand order repaired and checked by reading, not compiled, not run on hardware
//
#include <hip/hip_runtime.h>
#include <hip/hip_bf16.h>
#include <math.h>

// ---------------------------------------------------------------- types
typedef __attribute__((ext_vector_type(16))) _Float16 v16h;
typedef __attribute__((ext_vector_type(8)))  _Float16 v8h;
typedef __attribute__((ext_vector_type(8)))  float    v8f;
typedef __attribute__((ext_vector_type(4)))  unsigned int u32x4;
typedef __attribute__((ext_vector_type(8)))  int      i32x8;
typedef __attribute__((ext_vector_type(4)))  int      i32x4;
typedef _Float16 half_t;

#define Bn  4
#define Cn  512
#define HCn 64
#define Nn  4096   // 64*64 pixels

// Build a v16h fragment from two 16-byte chunks (A/B WMMA fragment loads).
__device__ __forceinline__ v16h pack16(const half_t* p0, const half_t* p1) {
  v8h a = *(const v8h*)p0;
  v8h b = *(const v8h*)p1;
  v16h r;
#pragma unroll
  for (int i = 0; i < 8; ++i) { r[i] = a[i]; r[i + 8] = b[i]; }
  return r;
}

__device__ __forceinline__ v8f wmma32(v16h a, v16h b, v8f c) {
  return __builtin_amdgcn_wmma_f32_16x16x32_f16(false, a, false, b, (short)0, c,
                                                false, false);
}

__device__ __forceinline__ v8f vzero8() {
  v8f z = {0.f, 0.f, 0.f, 0.f, 0.f, 0.f, 0.f, 0.f};
  return z;
}

// ---------------------------------------------------------------- TDM helper
// 2D tensor tile -> LDS via Tensor Data Mover (D# per CDNA5 ISA ch.8).
// tile is tile_d1 rows x 64 elements (f16), row stride = stride0 elements.
__device__ __forceinline__ void tdm_load_2d(unsigned lds_addr,
                                            const half_t* gaddr,
                                            unsigned tensor_d0,
                                            unsigned tensor_d1,
                                            unsigned stride0,
                                            unsigned tile_d1) {
  const unsigned long long ga = (unsigned long long)(size_t)gaddr;
  u32x4 g0;
  g0[0] = 1u;                                     // count=1, user descriptor
  g0[1] = lds_addr;                               // LDS byte address
  g0[2] = (unsigned)(ga & 0xffffffffu);           // global_addr[31:0]
  g0[3] = (unsigned)((ga >> 32) & 0x01ffffffu)    // global_addr[56:32]
        | (2u << 30);                             // type = 2 ("image")
  i32x8 g1;
  g1[0] = (int)(1u << 16);                        // data_size=1 -> 2 bytes
  g1[1] = (int)((tensor_d0 & 0xffffu) << 16);     // tensor_dim0[15:0]
  g1[2] = (int)(((tensor_d0 >> 16) & 0xffffu) |
                ((tensor_d1 & 0xffffu) << 16));   // tensor_dim0 hi | dim1 lo
  g1[3] = (int)(((tensor_d1 >> 16) & 0xffffu) |
                (64u << 16));                     // tensor_dim1 hi | tile_dim0=64
  g1[4] = (int)(tile_d1 & 0xffffu);               // tile_dim1 (tile_dim2=0)
  g1[5] = (int)stride0;                           // tensor_dim0_stride[31:0]
  g1[6] = 0;                                      // stride hi | dim1_stride lo
  g1[7] = 0;
  i32x4 z = {0, 0, 0, 0};
#if __clang_major__ >= 23
  i32x8 z8 = {0, 0, 0, 0, 0, 0, 0, 0};
  __builtin_amdgcn_tensor_load_to_lds(g0, g1, z, z, z8, 0);
#else
  __builtin_amdgcn_tensor_load_to_lds(g0, g1, z, z, 0);
#endif
}

// ---------------------------------------------------------------- spectral norm
// sigma = || W @ normalize(W^T @ u) || ;  out = (half)(W / sigma)
__global__ __launch_bounds__(256) void sn_kernel(
    const float* Wf, const float* uf, const float* Wg, const float* ug,
    const float* Wh, const float* uh, const float* Wv, const float* uv,
    half_t* of, half_t* og, half_t* oh, half_t* ov) {
  const float* W; const float* u; half_t* o; int R, Cc;
  switch (blockIdx.x) {
    case 0:  W = Wf; u = uf; o = of; R = HCn; Cc = Cn; break;
    case 1:  W = Wg; u = ug; o = og; R = HCn; Cc = Cn; break;
    case 2:  W = Wh; u = uh; o = oh; R = HCn; Cc = Cn; break;
    default: W = Wv; u = uv; o = ov; R = Cn; Cc = HCn; break;
  }
  __shared__ float v[Cn];
  __shared__ float red[256];
  const int t = threadIdx.x;

  for (int j = t; j < Cc; j += 256) {            // v = W^T u
    float s = 0.f;
    for (int i = 0; i < R; ++i) s += W[i * Cc + j] * u[i];
    v[j] = s;
  }
  __syncthreads();
  float ss = 0.f;
  for (int j = t; j < Cc; j += 256) ss += v[j] * v[j];
  red[t] = ss; __syncthreads();
  for (int s2 = 128; s2 > 0; s2 >>= 1) {
    if (t < s2) red[t] += red[t + s2];
    __syncthreads();
  }
  const float vn = fmaxf(sqrtf(red[0]), 1e-12f);
  __syncthreads();
  for (int j = t; j < Cc; j += 256) v[j] /= vn;
  __syncthreads();

  float ss2 = 0.f;                               // sigma = || W v ||
  for (int i = t; i < R; i += 256) {
    float s = 0.f;
    for (int j = 0; j < Cc; ++j) s += W[i * Cc + j] * v[j];
    ss2 += s * s;
  }
  red[t] = ss2; __syncthreads();
  for (int s2 = 128; s2 > 0; s2 >>= 1) {
    if (t < s2) red[t] += red[t + s2];
    __syncthreads();
  }
  const float inv = 1.0f / fmaxf(sqrtf(red[0]), 1e-20f);
  for (int e = t; e < R * Cc; e += 256) o[e] = (half_t)(W[e] * inv);
}

// ---------------------------------------------------------------- f/g/h projection
// f = Wf x + bf etc.  Writes fT, gT as [B][N][HC] (n-major) and h as [B][HC][N].
__global__ __launch_bounds__(128) void proj_kernel(
    const float* __restrict__ x,
    const half_t* __restrict__ Wf16, const float* __restrict__ bf,
    const half_t* __restrict__ Wg16, const float* __restrict__ bg,
    const half_t* __restrict__ Wh16, const float* __restrict__ bh,
    half_t* __restrict__ fT, half_t* __restrict__ gT, half_t* __restrict__ hO) {
  const int b  = blockIdx.y;
  const int n0 = blockIdx.x * 64;
  const int tid = threadIdx.x;
  const int lane = tid & 31, w = tid >> 5;
  const int lo = lane & 15, hf = lane >> 4;

  __shared__ alignas(16) half_t xT[64][48];  // 64 n x 32 c (f16), 96B rows
  const float* xb = x + (size_t)b * Cn * Nn;

  v8f acc[3][4];
#pragma unroll
  for (int m = 0; m < 3; ++m)
#pragma unroll
    for (int tt = 0; tt < 4; ++tt) acc[m][tt] = vzero8();

  const half_t* Ws[3] = {Wf16, Wg16, Wh16};

  for (int c0 = 0; c0 < Cn; c0 += 32) {
    // stage x chunk transposed to f16: xT[n][c]
    for (int e = tid; e < 32 * 64; e += 128) {
      const int cc = e >> 6, nn = e & 63;
      xT[nn][cc] = (half_t)xb[(size_t)(c0 + cc) * Nn + n0 + nn];
    }
    __syncthreads();

    const int koff = hf ? 16 : 0;
    const v16h Bx = pack16(&xT[w * 16 + lo][koff / 2], &xT[w * 16 + lo][koff / 2 + 16]);
    const int kb = hf ? 8 : 0;

    // issue all A-fragment loads first, then the WMMAs (better overlap)
    v16h A[3][4];
#pragma unroll
    for (int mat = 0; mat < 3; ++mat)
#pragma unroll
      for (int tt = 0; tt < 4; ++tt) {
        const half_t* row = Ws[mat] + (size_t)(tt * 16 + lo) * Cn + c0;
        A[mat][tt] = pack16(row + kb, row + kb + 16);
      }
#pragma unroll
    for (int mat = 0; mat < 3; ++mat)
#pragma unroll
      for (int tt = 0; tt < 4; ++tt)
        acc[mat][tt] = wmma32(A[mat][tt], Bx, acc[mat][tt]);
    __syncthreads();
  }

  // epilogue: bias + store in fragment-friendly layouts
  const int ncol = n0 + w * 16 + lo;
  const float* biases[3] = {bf, bg, bh};
#pragma unroll
  for (int mat = 0; mat < 3; ++mat) {
#pragma unroll
    for (int tt = 0; tt < 4; ++tt) {
      const int hcbase = tt * 16 + hf * 8;
      v8h outv;
#pragma unroll
      for (int r = 0; r < 8; ++r)
        outv[r] = (half_t)(acc[mat][tt][r] + biases[mat][hcbase + r]);
      if (mat == 0) {
        *(v8h*)(fT + ((size_t)b * Nn + ncol) * HCn + hcbase) = outv;
      } else if (mat == 1) {
        *(v8h*)(gT + ((size_t)b * Nn + ncol) * HCn + hcbase) = outv;
      } else {
#pragma unroll
        for (int r = 0; r < 8; ++r)
          hO[((size_t)b * HCn + hcbase + r) * Nn + ncol] = outv[r];
      }
    }
  }
}

// ---------------------------------------------------------------- flash attention
// beta = softmax(fT g); o = beta h^T -> oT [B][N][HC].
// Key/value tiles are DMA'd to LDS by the Tensor Data Mover, double-buffered.
__global__ __launch_bounds__(128) void attn_kernel(
    const half_t* __restrict__ fT, const half_t* __restrict__ gT,
    const half_t* __restrict__ hO, half_t* __restrict__ oT) {
  const int b  = blockIdx.y;
  const int n0 = blockIdx.x * 64;
  const int tid = threadIdx.x;
  const int lane = tid & 31, w = tid >> 5;
  const int lo = lane & 15, hf = lane >> 4;

  __shared__ alignas(16) half_t gbuf[2][64][64];  // keys:   [m_local][hc]
  __shared__ alignas(16) half_t hbuf[2][64][64];  // values: [hc][m_local]
  __shared__ alignas(16) half_t Pbuf[4][16][64];  // per-wave P strip

  const half_t* fTb = fT + (size_t)b * Nn * HCn;
  const half_t* gTb = gT + (size_t)b * Nn * HCn;
  const half_t* hb  = hO + (size_t)b * HCn * Nn;

  const int nw = n0 + w * 16;  // this wave's 16 query rows

  // persistent A fragments for f (K = HC = 64 -> two K=32 fragments)
  const half_t* frow = fTb + (size_t)(nw + lo) * HCn;
  const int kb = hf ? 8 : 0;
  const v16h Af0 = pack16(frow + kb, frow + kb + 16);
  const v16h Af1 = pack16(frow + 32 + kb, frow + 32 + kb + 16);

  v8f O[4];
#pragma unroll
  for (int t = 0; t < 4; ++t) O[t] = vzero8();
  float mrow[8], lrow[8];
#pragma unroll
  for (int r = 0; r < 8; ++r) { mrow[r] = -__builtin_inff(); lrow[r] = 0.f; }

  const int nstages = Nn / 64;

  // each wave DMAs a 16-row slice of the g-tile and of the h-tile
  auto issue_stage = [&](int stage, int buf) {
    const int m0 = stage * 64;
    tdm_load_2d((unsigned)(size_t)&gbuf[buf][w * 16][0],
                gTb + (size_t)(m0 + w * 16) * HCn,
                HCn, Nn, HCn, 16);
    tdm_load_2d((unsigned)(size_t)&hbuf[buf][w * 16][0],
                hb + (size_t)(w * 16) * Nn + m0,
                Nn, HCn, Nn, 16);
  };

  issue_stage(0, 0);

  for (int s = 0; s < nstages; ++s) {
    const int buf = s & 1;
    if (s + 1 < nstages) {
      issue_stage(s + 1, buf ^ 1);                 // prefetch next tile
      __builtin_amdgcn_s_wait_tensorcnt(2);        // current tile landed
    } else {
      __builtin_amdgcn_s_wait_tensorcnt(0);
    }
    __syncthreads();                               // publish LDS tile

    // ---- S strip: 16 queries x 64 keys (B fragments from LDS)
    v8f S[4];
    const int ko = hf ? 8 : 0;
#pragma unroll
    for (int j = 0; j < 4; ++j) {
      const half_t* grow = &gbuf[buf][j * 16 + lo][0];
      const v16h Bg0 = pack16(grow + ko, grow + ko + 16);
      const v16h Bg1 = pack16(grow + 32 + ko, grow + 32 + ko + 16);
      v8f c = vzero8();
      c = wmma32(Af0, Bg0, c);
      c = wmma32(Af1, Bg1, c);
      S[j] = c;
    }

    // ---- online softmax row stats (row r lives in VGPR r across a 16-lane half)
    float mnew[8], scale[8], psum[8];
#pragma unroll
    for (int r = 0; r < 8; ++r) {
      float mx = fmaxf(fmaxf(S[0][r], S[1][r]), fmaxf(S[2][r], S[3][r]));
#pragma unroll
      for (int off = 1; off < 16; off <<= 1) mx = fmaxf(mx, __shfl_xor(mx, off, 16));
      mnew[r]  = fmaxf(mrow[r], mx);
      scale[r] = __expf(mrow[r] - mnew[r]);
      mrow[r]  = mnew[r];
      psum[r]  = 0.f;
    }

    // ---- P = exp(S - m), spill to LDS (A-layout reload), partial sums
#pragma unroll
    for (int j = 0; j < 4; ++j) {
#pragma unroll
      for (int r = 0; r < 8; ++r) {
        const float p = __expf(S[j][r] - mnew[r]);
        psum[r] += p;
        Pbuf[w][r + 8 * hf][j * 16 + lo] = (half_t)p;
      }
    }
#pragma unroll
    for (int r = 0; r < 8; ++r) {
      float sm = psum[r];
#pragma unroll
      for (int off = 1; off < 16; off <<= 1) sm += __shfl_xor(sm, off, 16);
      lrow[r] = lrow[r] * scale[r] + sm;
    }

    // ---- rescale accumulators
#pragma unroll
    for (int t = 0; t < 4; ++t)
#pragma unroll
      for (int r = 0; r < 8; ++r) O[t][r] *= scale[r];

    // ---- reload P as A fragments (same-wave LDS is in-order)
    const half_t* prow = &Pbuf[w][lo][0];
    const int kb2 = hf ? 8 : 0;
    const v16h Ap0 = pack16(prow + kb2, prow + kb2 + 16);
    const v16h Ap1 = pack16(prow + 32 + kb2, prow + 32 + kb2 + 16);

    // ---- O += P @ h^T  (B fragments from LDS value tile)
#pragma unroll
    for (int t = 0; t < 4; ++t) {
      const half_t* hrow = &hbuf[buf][t * 16 + lo][0];
      const v16h Bh0 = pack16(hrow + ko, hrow + ko + 16);
      const v16h Bh1 = pack16(hrow + 32 + ko, hrow + 32 + ko + 16);
      O[t] = wmma32(Ap0, Bh0, O[t]);
      O[t] = wmma32(Ap1, Bh1, O[t]);
    }
    __syncthreads();  // all waves done with buf before it is overwritten
  }

  // ---- finalize: divide by row sums, store oT[b][n][hc]
  half_t* ob = oT + (size_t)b * Nn * HCn;
#pragma unroll
  for (int r = 0; r < 8; ++r) {
    const float inv = 1.0f / lrow[r];
#pragma unroll
    for (int t = 0; t < 4; ++t)
      ob[(size_t)(nw + r + 8 * hf) * HCn + t * 16 + lo] = (half_t)(O[t][r] * inv);
  }
}

// ---------------------------------------------------------------- output projection
// out = gamma * (Wv o + bv) + x
__global__ __launch_bounds__(128) void outproj_kernel(
    const half_t* __restrict__ Wv16, const float* __restrict__ bv,
    const half_t* __restrict__ oT, const float* __restrict__ x,
    const float* __restrict__ gamma, float* __restrict__ out) {
  const int b  = blockIdx.z;
  const int n0 = blockIdx.y * 64;
  const int c0 = blockIdx.x * 128;
  const int tid = threadIdx.x;
  const int lane = tid & 31, w = tid >> 5;
  const int lo = lane & 15, hf = lane >> 4;

  const half_t* oTb = oT + (size_t)b * Nn * HCn;

  v8f acc[2][4];
#pragma unroll
  for (int ct = 0; ct < 2; ++ct)
#pragma unroll
    for (int nt = 0; nt < 4; ++nt) acc[ct][nt] = vzero8();

  const int kb = hf ? 8 : 0;
  const int ko = hf ? 8 : 0;
#pragma unroll
  for (int k0 = 0; k0 < HCn; k0 += 32) {
    v16h A[2], Bv[4];
#pragma unroll
    for (int ct = 0; ct < 2; ++ct) {
      const half_t* row = Wv16 + (size_t)(c0 + (w * 2 + ct) * 16 + lo) * HCn + k0;
      A[ct] = pack16(row + kb, row + kb + 16);
    }
#pragma unroll
    for (int nt = 0; nt < 4; ++nt) {
      const half_t* col = oTb + (size_t)(n0 + nt * 16 + lo) * HCn + k0;
      Bv[nt] = pack16(col + ko, col + ko + 16);
    }
#pragma unroll
    for (int ct = 0; ct < 2; ++ct)
#pragma unroll
      for (int nt = 0; nt < 4; ++nt) acc[ct][nt] = wmma32(A[ct], Bv[nt], acc[ct][nt]);
  }

  const float gm = gamma[0];
#pragma unroll
  for (int ct = 0; ct < 2; ++ct) {
#pragma unroll
    for (int nt = 0; nt < 4; ++nt) {
#pragma unroll
      for (int r = 0; r < 8; ++r) {
        const int c = c0 + (w * 2 + ct) * 16 + r + 8 * hf;
        const int n = n0 + nt * 16 + lo;
        const size_t idx = ((size_t)b * Cn + c) * Nn + n;
        out[idx] = gm * (acc[ct][nt][r] + bv[c]) + x[idx];
      }
    }
  }
}

// ---------------------------------------------------------------- launch
extern "C" void kernel_launch(void* const* d_in, const int* in_sizes, int n_in,
                              void* d_out, int out_size, void* d_ws, size_t ws_size,
                              hipStream_t stream) {
  const float* x  = (const float*)d_in[0];
  const float* Wf = (const float*)d_in[1];
  const float* bf = (const float*)d_in[2];
  const float* Wg = (const float*)d_in[3];
  const float* bg = (const float*)d_in[4];
  const float* Wh = (const float*)d_in[5];
  const float* bh = (const float*)d_in[6];
  const float* Wv = (const float*)d_in[7];
  const float* bv = (const float*)d_in[8];
  const float* uf = (const float*)d_in[9];
  const float* ug = (const float*)d_in[10];
  const float* uh = (const float*)d_in[11];
  const float* uv = (const float*)d_in[12];
  const float* gamma = (const float*)d_in[13];
  float* out = (float*)d_out;

  // workspace carve-up (~8.25 MB)
  char* ws = (char*)d_ws;
  half_t* Wf16 = (half_t*)ws; ws += (size_t)HCn * Cn * sizeof(half_t);
  half_t* Wg16 = (half_t*)ws; ws += (size_t)HCn * Cn * sizeof(half_t);
  half_t* Wh16 = (half_t*)ws; ws += (size_t)HCn * Cn * sizeof(half_t);
  half_t* Wv16 = (half_t*)ws; ws += (size_t)Cn * HCn * sizeof(half_t);
  half_t* fT   = (half_t*)ws; ws += (size_t)Bn * Nn * HCn * sizeof(half_t);
  half_t* gT   = (half_t*)ws; ws += (size_t)Bn * Nn * HCn * sizeof(half_t);
  half_t* hBuf = (half_t*)ws; ws += (size_t)Bn * HCn * Nn * sizeof(half_t);
  half_t* oTb  = (half_t*)ws; ws += (size_t)Bn * Nn * HCn * sizeof(half_t);

  sn_kernel<<<4, 256, 0, stream>>>(Wf, uf, Wg, ug, Wh, uh, Wv, uv,
                                   Wf16, Wg16, Wh16, Wv16);
  proj_kernel<<<dim3(Nn / 64, Bn), 128, 0, stream>>>(x, Wf16, bf, Wg16, bg,
                                                     Wh16, bh, fT, gT, hBuf);
  attn_kernel<<<dim3(Nn / 64, Bn), 128, 0, stream>>>(fT, gT, hBuf, oTb);
  outproj_kernel<<<dim3(Cn / 128, Nn / 64, Bn), 128, 0, stream>>>(
      Wv16, bv, oTb, x, gamma, out);
}
